// MultiheadAttention_77601469104404
// MI455X (gfx1250) — matrix operand order repaired and checked by reading, not compiled, not run on hardware
//
#include <hip/hip_runtime.h>
#include <hip/hip_bf16.h>
#include <stdint.h>
#include <stddef.h>

// Problem constants (match reference)
#define B_  4
#define S_  2048
#define D_  1024
#define H_  16
#define DH_ 64
#define BS_ (B_ * S_)
#define BH_ (B_ * H_)

typedef __attribute__((ext_vector_type(16))) __bf16 v16bf;
typedef __attribute__((ext_vector_type(8)))  float  v8f;

// ---------- helpers ----------

__device__ __forceinline__ __bf16 f2bf(float f) {
    union { float f; uint32_t u; } v; v.f = f;
    uint32_t r = v.u + 0x7FFFu + ((v.u >> 16) & 1u);   // round-to-nearest-even
    union { uint16_t u; __bf16 b; } o; o.u = (uint16_t)(r >> 16);
    return o.b;
}

__device__ __forceinline__ uint32_t pack2bf(float lo, float hi) {
    union { __bf16 b[2]; uint32_t u; } o;
    o.b[0] = f2bf(lo); o.b[1] = f2bf(hi);
    return o.u;
}

// convert 8 consecutive f32 -> 8 bf16, 128-bit load x2, 128-bit store
__device__ __forceinline__ void cvt8(const float* __restrict__ src,
                                     __bf16* __restrict__ dst, size_t base) {
    const float4 x0 = *(const float4*)(src + base);
    const float4 x1 = *(const float4*)(src + base + 4);
    uint4 o;
    o.x = pack2bf(x0.x, x0.y); o.y = pack2bf(x0.z, x0.w);
    o.z = pack2bf(x1.x, x1.y); o.w = pack2bf(x1.z, x1.w);
    *(uint4*)(dst + base) = o;
}

__device__ __forceinline__ v8f wmma_bf16(v16bf a, v16bf b, v8f c) {
    // D = A(16x32 bf16) x B(32x16 bf16) + C(16x16 f32)
    return __builtin_amdgcn_wmma_f32_16x16x32_bf16(
        /*neg_a=*/false, a, /*neg_b=*/false, b,
        /*c_mod=*/(short)0, c, /*reuse_a=*/false, /*reuse_b=*/false);
}

// A-matrix 16x32 bf16 fragment: lane l -> row m=l%16;
// lane holds K = hb..hb+15, hb = 16*(l/16) (matches B).
__device__ __forceinline__ v16bf load_a_bf16(const __bf16* __restrict__ p, int ld) {
    const int lane = threadIdx.x & 31;
    const int m  = lane & 15;
    const int hb = (lane >> 4) * 16;
    const __bf16* row = p + (size_t)m * ld;
    v16bf a;
#pragma unroll
    for (int i = 0; i < 16; ++i) a[i] = row[hb + i];
    return a;
}

// B-matrix 32x16 bf16 fragment: lane l -> col n=l%16, K = 16*(l/16)+0..15.
// Element (k,n) at p[n*sn + k]; 16 contiguous bf16 per lane.
__device__ __forceinline__ v16bf load_b_bf16(const __bf16* __restrict__ p, int sn) {
    const int lane = threadIdx.x & 31;
    const int n  = lane & 15;
    const int kb = (lane >> 4) * 16;
    const __bf16* row = p + (size_t)n * sn + kb;
    v16bf b;
#pragma unroll
    for (int i = 0; i < 16; ++i) b[i] = row[i];
    return b;
}

// ---------- kernel 1: f32 -> bf16 conversion (3 arrays, 8 elems/thread) ----------

__global__ void cvt3_kernel(const float* __restrict__ a0, const float* __restrict__ a1,
                            const float* __restrict__ a2,
                            __bf16* __restrict__ o0, __bf16* __restrict__ o1,
                            __bf16* __restrict__ o2, size_t n8) {
    const size_t i = (size_t)blockIdx.x * blockDim.x + threadIdx.x;
    if (i >= n8) return;
    const size_t base = i * 8;
    cvt8(a0, o0, base);
    cvt8(a1, o1, base);
    cvt8(a2, o2, base);
}

// ---------- kernel 2: projection GEMM  y = x @ W^T + b ----------
// Each wave: 16x64 output tile (one full head width). 4 WMMA accs.
// mode 0: Q -> bf16 (b,h,s,dh), pre-scaled by 1/sqrt(DH)=0.125
// mode 1: K -> bf16 (b,h,s,dh)
// mode 2: V -> bf16 transposed (b,h,dh,s)
__global__ void proj_kernel(const __bf16* __restrict__ xb,    // (B*S, D) bf16
                            const __bf16* __restrict__ Wb,    // (D, D) bf16
                            const float*  __restrict__ bias,  // (D,) f32
                            __bf16* __restrict__ outb,
                            int mode) {
    const int lane  = threadIdx.x & 31;
    const int wave  = threadIdx.x >> 5;
    const int mtile = blockIdx.x * 8 + wave;   // 0..511
    const int h     = blockIdx.y;              // 0..15 (head = 64-wide N group)
    const int m0 = mtile * 16, n0 = h * 64;

    v8f acc[4] = {v8f{}, v8f{}, v8f{}, v8f{}};
    for (int kk = 0; kk < D_; kk += 32) {
        const v16bf a = load_a_bf16(xb + (size_t)m0 * D_ + kk, D_);
#pragma unroll
        for (int s = 0; s < 4; ++s) {
            const v16bf b = load_b_bf16(Wb + (size_t)(n0 + s * 16) * D_ + kk, D_);
            acc[s] = wmma_bf16(a, b, acc[s]);
        }
    }

    const int n  = lane & 15;
    const int hf = lane >> 4;
#pragma unroll
    for (int s = 0; s < 4; ++s) {
        const int dh  = s * 16 + n;            // 0..63 within head
        const float bv = bias[n0 + dh];
#pragma unroll
        for (int r = 0; r < 8; ++r) {
            const int m  = m0 + r + hf * 8;    // global row in B*S
            const int bb = m >> 11;            // m / S_
            const int sr = m & 2047;           // m % S_
            const float vv = acc[s][r] + bv;
            if (mode == 0) {
                outb[((size_t)(bb * H_ + h) * S_ + sr) * DH_ + dh] = f2bf(vv * 0.125f);
            } else if (mode == 1) {
                outb[((size_t)(bb * H_ + h) * S_ + sr) * DH_ + dh] = f2bf(vv);
            } else {
                outb[((size_t)(bb * H_ + h) * DH_ + dh) * S_ + sr] = f2bf(vv);
            }
        }
    }
}

// ---------- kernel 3: scores = Q_s @ K^T  (per b,h), 16x64 per wave ----------
__global__ void scores_kernel(const __bf16* __restrict__ Qb,   // (BH, S, DH)
                              const __bf16* __restrict__ Kb,   // (BH, S, DH)
                              float* __restrict__ attnF) {     // (BH, S, S) in d_out
    const int lane  = threadIdx.x & 31;
    const int wave  = threadIdx.x >> 5;
    const int mtile = blockIdx.x * 8 + wave;   // 0..127
    const int ng    = blockIdx.y;              // 0..31 (64-wide N group)
    const int bh    = blockIdx.z;              // 0..63
    const int m0 = mtile * 16, n0 = ng * 64;

    const __bf16* Qbase = Qb + (size_t)bh * S_ * DH_;
    const __bf16* Kbase = Kb + (size_t)bh * S_ * DH_;

    v8f acc[4] = {v8f{}, v8f{}, v8f{}, v8f{}};
#pragma unroll
    for (int kk = 0; kk < DH_; kk += 32) {
        const v16bf a = load_a_bf16(Qbase + (size_t)m0 * DH_ + kk, DH_);
#pragma unroll
        for (int s = 0; s < 4; ++s) {
            const v16bf b = load_b_bf16(Kbase + (size_t)(n0 + s * 16) * DH_ + kk, DH_);
            acc[s] = wmma_bf16(a, b, acc[s]);
        }
    }

    const int n  = lane & 15;
    const int hf = lane >> 4;
    float* obase = attnF + (size_t)bh * S_ * S_;
#pragma unroll
    for (int s = 0; s < 4; ++s) {
#pragma unroll
        for (int r = 0; r < 8; ++r) {
            obase[(size_t)(m0 + r + hf * 8) * S_ + n0 + s * 16 + n] = acc[s][r];
        }
    }
}

// ---------- kernel 4: row softmax, in-place f32 + packed bf16 copy ----------
__global__ void softmax_kernel(float* __restrict__ attnF,      // (BH*S, S)
                               __bf16* __restrict__ attnB) {   // (BH*S, S)
    const size_t row = blockIdx.x;
    float*  p  = attnF + row * (size_t)S_;
    __bf16* pb = attnB + row * (size_t)S_;
    const int t = threadIdx.x;                 // 0..255, 8 contiguous elems each

    __shared__ float red[256];
    const float4 x0 = *(const float4*)(p + t * 8);
    const float4 x1 = *(const float4*)(p + t * 8 + 4);
    float vals[8] = {x0.x, x0.y, x0.z, x0.w, x1.x, x1.y, x1.z, x1.w};

    float lmax = vals[0];
#pragma unroll
    for (int i = 1; i < 8; ++i) lmax = fmaxf(lmax, vals[i]);
    red[t] = lmax; __syncthreads();
#pragma unroll
    for (int s2 = 128; s2 > 0; s2 >>= 1) {
        if (t < s2) red[t] = fmaxf(red[t], red[t + s2]);
        __syncthreads();
    }
    const float rmax = red[0];
    __syncthreads();

    float lsum = 0.0f;
#pragma unroll
    for (int i = 0; i < 8; ++i) {
        vals[i] = __expf(vals[i] - rmax);
        lsum += vals[i];
    }
    red[t] = lsum; __syncthreads();
#pragma unroll
    for (int s2 = 128; s2 > 0; s2 >>= 1) {
        if (t < s2) red[t] += red[t + s2];
        __syncthreads();
    }
    const float inv = 1.0f / red[0];

#pragma unroll
    for (int i = 0; i < 8; ++i) vals[i] *= inv;
    *(float4*)(p + t * 8)     = make_float4(vals[0], vals[1], vals[2], vals[3]);
    *(float4*)(p + t * 8 + 4) = make_float4(vals[4], vals[5], vals[6], vals[7]);
    uint4 ob;
    ob.x = pack2bf(vals[0], vals[1]); ob.y = pack2bf(vals[2], vals[3]);
    ob.z = pack2bf(vals[4], vals[5]); ob.w = pack2bf(vals[6], vals[7]);
    *(uint4*)(pb + t * 8) = ob;
}

// ---------- kernel 5: Hout = attn @ V  (per b,h), 16x64 per wave ----------
// attnB (536 MB) is read exactly once; prefetch ahead on the stream.
__global__ void out_kernel(const __bf16* __restrict__ attnB,   // (BH, S, S)
                           const __bf16* __restrict__ VT,      // (BH, DH, S)
                           float* __restrict__ Hout) {         // (B, S, D) in d_out
    const int lane  = threadIdx.x & 31;
    const int wave  = threadIdx.x >> 5;
    const int mtile = blockIdx.x * 8 + wave;   // 0..127
    const int bh    = blockIdx.z;              // 0..63
    const int m0 = mtile * 16;

    const __bf16* Abase = attnB + (size_t)bh * S_ * S_ + (size_t)m0 * S_;
    const __bf16* Vbase = VT    + (size_t)bh * DH_ * S_;
    const __bf16* arow  = Abase + (size_t)(lane & 15) * S_;

    v8f acc[4] = {v8f{}, v8f{}, v8f{}, v8f{}};
    for (int kk = 0; kk < S_; kk += 32) {
        // stream-ahead prefetch of the one-pass attn operand (global_prefetch_b8)
        __builtin_prefetch(arow + kk + 512, 0, 0);
        const v16bf a = load_a_bf16(Abase + kk, S_);
#pragma unroll
        for (int s = 0; s < 4; ++s) {
            const v16bf b = load_b_bf16(Vbase + (size_t)(s * 16) * S_ + kk, S_);
            acc[s] = wmma_bf16(a, b, acc[s]);
        }
    }

    const int n  = lane & 15;
    const int hf = lane >> 4;
    const int bb = bh / H_;
    const int h  = bh % H_;
#pragma unroll
    for (int s = 0; s < 4; ++s) {
#pragma unroll
        for (int r = 0; r < 8; ++r) {
            const int sr = m0 + r + hf * 8;
            Hout[((size_t)bb * S_ + sr) * D_ + h * DH_ + s * 16 + n] = acc[s][r];
        }
    }
}

// ---------- host launcher ----------

extern "C" void kernel_launch(void* const* d_in, const int* in_sizes, int n_in,
                              void* d_out, int out_size, void* d_ws, size_t ws_size,
                              hipStream_t stream) {
    const float* q    = (const float*)d_in[0];
    const float* k    = (const float*)d_in[1];
    const float* v    = (const float*)d_in[2];
    const float* Wq_w = (const float*)d_in[3];
    const float* Wq_b = (const float*)d_in[4];
    const float* Wk_w = (const float*)d_in[5];
    const float* Wk_b = (const float*)d_in[6];
    const float* Wv_w = (const float*)d_in[7];
    const float* Wv_b = (const float*)d_in[8];

    float* Hout  = (float*)d_out;                                    // B*S*D
    float* attnF = (float*)d_out + (size_t)B_ * S_ * D_;             // BH*S*S

    // workspace layout (bytes)
    char* ws = (char*)d_ws;
    const size_t szW  = (size_t)D_ * D_ * sizeof(__bf16);            // 2 MB each
    const size_t szX  = (size_t)BS_ * D_ * sizeof(__bf16);           // 16 MB each
    const size_t szP  = (size_t)BH_ * S_ * DH_ * sizeof(__bf16);     // 16 MB each
    __bf16* Wqb   = (__bf16*)(ws);
    __bf16* Wkb   = (__bf16*)(ws + szW);
    __bf16* Wvb   = (__bf16*)(ws + 2 * szW);
    __bf16* xqb   = (__bf16*)(ws + 3 * szW);
    __bf16* xkb   = (__bf16*)(ws + 3 * szW + szX);
    __bf16* xvb   = (__bf16*)(ws + 3 * szW + 2 * szX);
    __bf16* Qb    = (__bf16*)(ws + 3 * szW + 3 * szX);
    __bf16* Kb    = (__bf16*)(ws + 3 * szW + 3 * szX + szP);
    __bf16* VTb   = (__bf16*)(ws + 3 * szW + 3 * szX + 2 * szP);
    __bf16* attnB = (__bf16*)(ws + 3 * szW + 3 * szX + 3 * szP);     // 512 MB

    // 1) convert weights and activations to bf16 (packed 128-bit I/O)
    {
        const size_t n8w = (size_t)D_ * D_ / 8;
        cvt3_kernel<<<dim3((unsigned)((n8w + 255) / 256)), dim3(256), 0, stream>>>(
            Wq_w, Wk_w, Wv_w, Wqb, Wkb, Wvb, n8w);
        const size_t n8x = (size_t)BS_ * D_ / 8;
        cvt3_kernel<<<dim3((unsigned)((n8x + 255) / 256)), dim3(256), 0, stream>>>(
            q, k, v, xqb, xkb, xvb, n8x);
    }
    // 2) projections: 512 M-tiles x 16 head-groups, 8 waves/block, 16x64/wave
    {
        dim3 grid(64, 16, 1), block(256, 1, 1);
        proj_kernel<<<grid, block, 0, stream>>>(xqb, Wqb, Wq_b, Qb,  0);
        proj_kernel<<<grid, block, 0, stream>>>(xkb, Wkb, Wk_b, Kb,  1);
        proj_kernel<<<grid, block, 0, stream>>>(xvb, Wvb, Wv_b, VTb, 2);
    }
    // 3) scores -> d_out attn region (raw), 16x64 per wave
    {
        dim3 grid(16, 32, 64), block(256, 1, 1);
        scores_kernel<<<grid, block, 0, stream>>>(Qb, Kb, attnF);
    }
    // 4) softmax in place + bf16 copy
    {
        softmax_kernel<<<dim3(BH_ * S_), dim3(256), 0, stream>>>(attnF, attnB);
    }
    // 5) attn @ V -> Hout, 16x64 per wave (full DH), single pass over attnB
    {
        dim3 grid(16, 1, 64), block(256, 1, 1);
        out_kernel<<<grid, block, 0, stream>>>(attnB, VTb, Hout);
    }
}
